// LePEAttention_47493748359184
// MI455X (gfx1250) — matrix operand order repaired and checked by reading, not compiled, not run on hardware
//
#include <hip/hip_runtime.h>
#include <cstdint>

#define DIMC   128
#define HEADS  8
#define HD     16
#define RESV   32
#define LTOK   (RESV*RESV*RESV)   // 32768
#define WIN    512
#define BATCH  2
#define QSCALE 0.25f              // HD^-0.5

typedef __attribute__((ext_vector_type(16))) _Float16 v16h;
typedef __attribute__((ext_vector_type(8)))  float    v8f;

union Frag {
  v16h  v;
  uint4 q[2];
};

__device__ __forceinline__ uint16_t f2h_bits(float f) {
  union { _Float16 h; uint16_t u; } cv;
  cv.h = (_Float16)f;
  return cv.u;
}

// LDS strides (elements): 16B-aligned rows + bank spreading
#define QS_STR 24    // f16 (16 used), 48B rows
#define KS_STR 40    // f16 (16 data + 16 zero pad + 8 slack), 80B rows
#define VT_STR 520   // f16, 1040B rows
#define SP_STR 520   // f32, 2080B rows (16B aligned -> float4 softmax)
#define PP_STR 520   // f16, 1040B rows

// byte offsets inside dynamic LDS
#define OFF_QS  0
#define OFF_KS  (512*QS_STR*2)                       // 24576
#define OFF_VT  (OFF_KS + 512*KS_STR*2)              // 65536
#define OFF_WV  (OFF_VT + 16*VT_STR*2)               // 82176
#define WAVE_SP (16*SP_STR*4)                        // 33280
#define WAVE_PP (16*PP_STR*2)                        // 16640
#define WAVE_SZ (WAVE_SP + WAVE_PP)                  // 49920
#define SHM_ATTN (OFF_WV + 4*WAVE_SZ)                // 281856 bytes < 320KB

__global__ __launch_bounds__(128)
void attn_kernel(const float* __restrict__ qkv, float* __restrict__ out) {
  extern __shared__ char smem[];
  uint16_t* qs = (uint16_t*)(smem + OFF_QS);
  uint16_t* ks = (uint16_t*)(smem + OFF_KS);
  uint16_t* vt = (uint16_t*)(smem + OFF_VT);

  const int tid  = threadIdx.x;
  const int wave = tid >> 5;
  const int lane = tid & 31;
  const bool lo  = lane < 16;
  const int mrow = lane & 15;

  const int blk = blockIdx.x;          // 1024 = 128 windows * 8 heads
  const int win = blk >> 3;
  const int h   = blk & 7;
  const int b   = win >> 6;
  const int iw  = (win >> 3) & 7;
  const int is  = win & 7;

  const float* qg = qkv + ((size_t)(0*BATCH + b) * LTOK) * DIMC;
  const float* kg = qkv + ((size_t)(1*BATCH + b) * LTOK) * DIMC;
  const float* vg = qkv + ((size_t)(2*BATCH + b) * LTOK) * DIMC;

  // zero the K padding region (contraction K=16..31 reads for the hi half-wave)
  for (int qi = tid; qi < WIN; qi += 128) {
    uint4 z = make_uint4(0u, 0u, 0u, 0u);
    *(uint4*)(ks + (size_t)qi*KS_STR + 16) = z;
    *(uint4*)(ks + (size_t)qi*KS_STR + 24) = z;
  }
  // stage Q (pre-scaled), K row-major [token][hd]; V transposed [hd][token]
  for (int idx = tid; idx < WIN*HD; idx += 128) {
    int d  = idx & 15;
    int qi = idx >> 4;
    int hs = qi >> 4, ws = (qi >> 2) & 3, ss = qi & 3;
    int l  = (hs << 10) | (((iw << 2) | ws) << 5) | ((is << 2) | ss);
    size_t g = (size_t)l * DIMC + h * HD + d;
    qs[qi*QS_STR + d] = f2h_bits(qg[g] * QSCALE);
    ks[qi*KS_STR + d] = f2h_bits(kg[g]);
    vt[d*VT_STR + qi] = f2h_bits(vg[g]);
  }
  __syncthreads();

  float*    sp = (float*)   (smem + OFF_WV + wave*WAVE_SZ);
  uint16_t* pp = (uint16_t*)(smem + OFF_WV + wave*WAVE_SZ + WAVE_SP);

  const uint16_t* krow = ks + (size_t)mrow*KS_STR + (lo ? 0 : 16);
  const uint16_t* vrow = vt + (size_t)mrow*VT_STR + (lo ? 0 : 8);
  const uint16_t* pqrow = pp + (size_t)mrow*PP_STR + (lo ? 0 : 8);

  for (int qt = 0; qt < 8; ++qt) {
    const int qbase = wave*128 + qt*16;

    // A fragment of Q tile: lanes 0-15 hold hd 0..15 (K=0..15);
    // lanes 16-31 (K=16..31) are zero padding (register constant).
    Frag aQ;
    {
      const uint4* p = (const uint4*)(qs + (size_t)(qbase + mrow)*QS_STR);
      aQ.q[0] = lo ? p[0] : make_uint4(0u, 0u, 0u, 0u);
      aQ.q[1] = lo ? p[1] : make_uint4(0u, 0u, 0u, 0u);
    }

    // ---- S = (Q*scale) K^T over 32 key tiles (double-buffered fragments) ----
    const int rb = lo ? 0 : 8;
    Frag bK;
    {
      const uint4* p = (const uint4*)krow;   // kt = 0
      bK.q[0] = p[0];
      bK.q[1] = p[1];
    }
    #pragma unroll
    for (int kt = 0; kt < 32; ++kt) {
      Frag bKn;
      if (kt < 31) {
        const uint4* p = (const uint4*)(krow + (size_t)(kt + 1)*16*KS_STR);
        bKn.q[0] = p[0];
        bKn.q[1] = p[1];
      }
      v8f c = {0.f,0.f,0.f,0.f,0.f,0.f,0.f,0.f};
      c = __builtin_amdgcn_wmma_f32_16x16x32_f16(false, aQ.v, false, bK.v,
                                                 (short)0, c, false, false);
      #pragma unroll
      for (int r = 0; r < 8; ++r)
        sp[(r + rb)*SP_STR + kt*16 + mrow] = c[r];
      bK = bKn;
    }

    // ---- softmax: row = mrow, each half-wave owns 256 columns (vectorized) ----
    const int cb = lo ? 0 : 256;
    const float4* srow = (const float4*)(sp + (size_t)mrow*SP_STR + cb);
    float m = -1e30f;
    for (int i = 0; i < 64; ++i) {
      float4 s = srow[i];
      m = fmaxf(m, fmaxf(fmaxf(s.x, s.y), fmaxf(s.z, s.w)));
    }
    m = fmaxf(m, __shfl_xor(m, 16));
    float sum = 0.f;
    uint2* prow = (uint2*)(pp + (size_t)mrow*PP_STR + cb);
    for (int i = 0; i < 64; ++i) {
      float4 s = srow[i];
      float e0 = __expf(s.x - m), e1 = __expf(s.y - m);
      float e2 = __expf(s.z - m), e3 = __expf(s.w - m);
      sum += (e0 + e1) + (e2 + e3);
      uint2 pk;
      pk.x = (uint32_t)f2h_bits(e0) | ((uint32_t)f2h_bits(e1) << 16);
      pk.y = (uint32_t)f2h_bits(e2) | ((uint32_t)f2h_bits(e3) << 16);
      prow[i] = pk;
    }
    sum += __shfl_xor(sum, 16);
    const float rinv = 1.0f / sum;

    // ---- O = P V, full K=32 contraction chunks (double-buffered fragments) ----
    v8f acc = {0.f,0.f,0.f,0.f,0.f,0.f,0.f,0.f};
    Frag aP, bV;
    {
      const uint4* ap = (const uint4*)pqrow;       // j0 = 0
      aP.q[0] = ap[0];
      aP.q[1] = ap[2];
      const uint4* bp = (const uint4*)vrow;
      bV.q[0] = bp[0];
      bV.q[1] = bp[2];
    }
    #pragma unroll
    for (int jt = 0; jt < 16; ++jt) {
      Frag aPn, bVn;
      if (jt < 15) {
        const uint4* ap = (const uint4*)(pqrow + (size_t)(jt + 1)*32);
        aPn.q[0] = ap[0];      // keys j0+base+0..7
        aPn.q[1] = ap[2];      // keys j0+base+16..23
        const uint4* bp = (const uint4*)(vrow + (size_t)(jt + 1)*32);
        bVn.q[0] = bp[0];
        bVn.q[1] = bp[2];
      }
      acc = __builtin_amdgcn_wmma_f32_16x16x32_f16(false, aP.v, false, bV.v,
                                                   (short)0, acc, false, false);
      aP = aPn;
      bV = bVn;
    }

    // ---- normalize and write to global output (windows2img layout) ----
    #pragma unroll
    for (int r = 0; r < 8; ++r) {
      float ri   = __shfl(rinv, r + rb);
      int   qrow = qbase + r + rb;
      int hs = qrow >> 4, ws = (qrow >> 2) & 3, ss = qrow & 3;
      int l  = (hs << 10) | (((iw << 2) | ws) << 5) | ((is << 2) | ss);
      out[((size_t)b * LTOK + l) * DIMC + h*HD + mrow] = acc[r] * ri;
    }
  }
}

// ---------------- LePE depthwise 5x5x5 conv (per window, zero-padded) ----------------
#define VB_STR 516
#define SHM_LEPE ((size_t)(32*VB_STR + 32*125 + 32) * 4)

__global__ __launch_bounds__(256)
void lepe_kernel(const float* __restrict__ qkv, const float* __restrict__ w,
                 const float* __restrict__ bias, float* __restrict__ out) {
  extern __shared__ char smem2[];
  float* vbuf = (float*)smem2;                      // [32][VB_STR]
  float* wbuf = (float*)(smem2 + 32*VB_STR*4);      // [32][125]
  float* bbuf = wbuf + 32*125;                      // [32]

  const int tid = threadIdx.x;
  const int blk = blockIdx.x;                       // 512 = 128 windows * 4 chan-groups
  const int win = blk >> 2;
  const int cbase = (blk & 3) * 32;
  const int b  = win >> 6;
  const int iw = (win >> 3) & 7;
  const int is = win & 7;
  const float* vg = qkv + ((size_t)(2*BATCH + b) * LTOK) * DIMC;

  for (int idx = tid; idx < 32*WIN; idx += 256) {
    int c  = idx & 31;
    int qi = idx >> 5;
    int hs = qi >> 4, ws = (qi >> 2) & 3, ss = qi & 3;
    int l  = (hs << 10) | (((iw << 2) | ws) << 5) | ((is << 2) | ss);
    vbuf[c*VB_STR + qi] = vg[(size_t)l * DIMC + cbase + c];
  }
  for (int idx = tid; idx < 32*125; idx += 256)
    wbuf[idx] = w[cbase*125 + idx];
  if (tid < 32) bbuf[tid] = bias[cbase + tid];
  __syncthreads();

  for (int k = 0; k < 64; ++k) {
    int oid = tid + k*256;
    int c   = oid & 31;
    int qi  = oid >> 5;
    int hs = qi >> 4, ws = (qi >> 2) & 3, ss = qi & 3;
    float acc = bbuf[c];
    const float* wc = wbuf + c*125;
    const float* vc = vbuf + c*VB_STR;
    #pragma unroll
    for (int dh = 0; dh < 5; ++dh) {
      int ih = hs + dh - 2;
      if ((unsigned)ih >= 32u) continue;
      #pragma unroll
      for (int dw = 0; dw < 5; ++dw) {
        int iw2 = ws + dw - 2;
        if ((unsigned)iw2 >= 4u) continue;
        #pragma unroll
        for (int ds = 0; ds < 5; ++ds) {
          int is2 = ss + ds - 2;
          if ((unsigned)is2 >= 4u) continue;
          acc += vc[(ih << 4) + (iw2 << 2) + is2] * wc[dh*25 + dw*5 + ds];
        }
      }
    }
    int l = (hs << 10) | (((iw << 2) | ws) << 5) | ((is << 2) | ss);
    out[((size_t)b * LTOK + l) * DIMC + cbase + c] += acc;   // attn kernel wrote first
  }
}

extern "C" void kernel_launch(void* const* d_in, const int* in_sizes, int n_in,
                              void* d_out, int out_size, void* d_ws, size_t ws_size,
                              hipStream_t stream) {
  (void)in_sizes; (void)n_in; (void)d_ws; (void)ws_size; (void)out_size;
  const float* qkv  = (const float*)d_in[0];
  const float* w    = (const float*)d_in[1];
  const float* bias = (const float*)d_in[2];
  float* out = (float*)d_out;

  hipFuncSetAttribute((const void*)attn_kernel,
                      hipFuncAttributeMaxDynamicSharedMemorySize, (int)SHM_ATTN);
  hipFuncSetAttribute((const void*)lepe_kernel,
                      hipFuncAttributeMaxDynamicSharedMemorySize, (int)SHM_LEPE);

  attn_kernel<<<dim3(1024), dim3(128), SHM_ATTN, stream>>>(qkv, out);
  lepe_kernel<<<dim3(512),  dim3(256), SHM_LEPE, stream>>>(qkv, w, bias, out);
}
